// LstmNet_28973849378857
// MI455X (gfx1250) — matrix operand order repaired and checked by reading, not compiled, not run on hardware
//
#include <hip/hip_runtime.h>
#include <hip/hip_bf16.h>

typedef __bf16 bf16_t;
typedef __attribute__((ext_vector_type(16))) __bf16 v16bf;
typedef __attribute__((ext_vector_type(8)))  __bf16 v8bf;
typedef __attribute__((ext_vector_type(8)))  float  v8f;
typedef __attribute__((ext_vector_type(4)))  unsigned int v4u;
typedef __attribute__((ext_vector_type(4)))  int v4i_t;
typedef __attribute__((ext_vector_type(8)))  int v8i_t;

#define T_STEPS 8192
#define HID     512
#define G4      2048   // 4*H

#if defined(__has_builtin)
#  if __has_builtin(__builtin_amdgcn_tensor_load_to_lds) && \
      __has_builtin(__builtin_amdgcn_s_wait_tensorcnt)
#    define HAVE_TDM 1
#  endif
#endif
#ifndef HAVE_TDM
#  define HAVE_TDM 0
#endif

// LDS panel row stride: 64 bf16 payload + 8 bf16 pad = 72 elem = 36 dwords.
// TDM produces this via pad_interval=4 (every 32 dwords) pad_amount=3 (4 dwords).
#define PANEL_STRIDE 72

// ---------------------------------------------------------------------------
// f32 -> bf16 conversion (grid-stride)
// ---------------------------------------------------------------------------
__global__ void f32_to_bf16_kernel(const float* __restrict__ in,
                                   bf16_t* __restrict__ out, int n) {
    int i = blockIdx.x * blockDim.x + threadIdx.x;
    int stride = gridDim.x * blockDim.x;
    for (; i < n; i += stride) out[i] = (bf16_t)in[i];
}

#if HAVE_TDM
// Issue one TDM descriptor: load a 64(rows) x 64(k) bf16 tile whose rows are
// HID apart in global memory into LDS at lds_addr, padded to PANEL_STRIDE.
// D# layout per CDNA5 ISA.
__device__ __forceinline__ void tdm_load_tile(unsigned lds_addr,
                                              const bf16_t* gptr,
                                              unsigned tensor_rows) {
    unsigned long long ga = (unsigned long long)(uintptr_t)gptr;
    v4u g0;
    g0[0] = 1u;                                     // count=1, user mode
    g0[1] = lds_addr;                               // LDS byte address
    g0[2] = (unsigned)(ga & 0xFFFFFFFFu);           // global_addr[31:0]
    g0[3] = (unsigned)((ga >> 32) & 0x01FFFFFFu)    // global_addr[56:32]
          | (2u << 30);                             // type=2 ("image")
    const unsigned td0 = HID;                       // tensor extent along k
    const unsigned td1 = tensor_rows;               // tensor extent along rows
    v8i_t g1;
    g1[0] = (int)((1u << 16)      // data_size = 1 -> 2 bytes
                | (1u << 20)      // pad_enable
                | (4u << 22)      // pad_interval: every 32 dwords
                | (3u << 25));    // pad_amount: 4 dwords
    g1[1] = (int)((td0 & 0xFFFFu) << 16);                    // tensor_dim0 lo
    g1[2] = (int)((td0 >> 16) | ((td1 & 0xFFFFu) << 16));    // td0 hi | td1 lo
    g1[3] = (int)((td1 >> 16) | (64u << 16));                // td1 hi | tile_dim0=64
    g1[4] = 64;                                              // tile_dim1=64, tile_dim2=0
    g1[5] = HID;                                             // tensor_dim0_stride=512
    g1[6] = 0;
    g1[7] = 0;
    v4i_t z4 = {0, 0, 0, 0};
#if __clang_major__ >= 23
    v8i_t z8 = {0, 0, 0, 0, 0, 0, 0, 0};
    __builtin_amdgcn_tensor_load_to_lds(g0, g1, z4, z4, z8, 0);
#else
    __builtin_amdgcn_tensor_load_to_lds(g0, g1, z4, z4, 0);
#endif
}
#endif

// ---------------------------------------------------------------------------
// C[M x 2048] = A[M x 512](bf16) @ W[2048 x 512]^T(bf16) + (b_ih + b_hh)
// Block: 64x64 output tile, K staged 64 at a time into LDS by the TDM
// (wave 0 issues tensor_load_to_lds; s_wait_tensorcnt + barrier publishes).
// 8 waves: wave w -> M-subtile (w&3), N-subtiles (w>>2)*2 + {0,1}; each
// subtile iterated with v_wmma_f32_16x16x32_bf16 from ds_load_b128 fragments.
// Fragment layouts per CDNA5 ISA 7.12.2.
// ---------------------------------------------------------------------------
__global__ void gemm_wmma_bf16_kernel(const bf16_t* __restrict__ A,
                                      const bf16_t* __restrict__ W,
                                      const float* __restrict__ b_ih,
                                      const float* __restrict__ b_hh,
                                      float* __restrict__ C, int M) {
    __shared__ bf16_t sA[64 * PANEL_STRIDE];   // 9 KB
    __shared__ bf16_t sB[64 * PANEL_STRIDE];   // 9 KB

    const int lane = threadIdx.x & 31;
    const int wave = threadIdx.x >> 5;
    const int tilesN = G4 / 64;                // 32
    const int bm = blockIdx.x / tilesN;
    const int bn = blockIdx.x % tilesN;
    const int wm = wave & 3;                   // M subtile 0..3
    const int wn = wave >> 2;                  // N subtile pair 0..1

    const int l15 = lane & 15;
    const int lhi = lane >> 4;

    v8f acc0 = {}, acc1 = {};

    for (int k0 = 0; k0 < HID; k0 += 64) {
        __syncthreads();                       // previous panels consumed
#if HAVE_TDM
        if (wave == 0) {
            tdm_load_tile((unsigned)(uintptr_t)&sA[0],
                          A + (size_t)bm * 64 * HID + k0, (unsigned)M);
            tdm_load_tile((unsigned)(uintptr_t)&sB[0],
                          W + (size_t)bn * 64 * HID + k0, (unsigned)G4);
            __builtin_amdgcn_s_wait_tensorcnt(0);
        }
#else
        for (int idx = threadIdx.x; idx < 64 * 64; idx += 256) {
            int r = idx >> 6, c = idx & 63;
            sA[r * PANEL_STRIDE + c] = A[((size_t)bm * 64 + r) * HID + k0 + c];
            sB[r * PANEL_STRIDE + c] = W[((size_t)bn * 64 + r) * HID + k0 + c];
        }
#endif
        __syncthreads();                       // panels visible to all waves

        #pragma unroll
        for (int kk = 0; kk < 64; kk += 32) {
            // A fragment: lane -> row M=l15, K = kk + lhi*8 + {0..7, 16..23}
            const bf16_t* pa = sA + (wm * 16 + l15) * PANEL_STRIDE + kk + lhi * 8;
            v8bf a0 = *(const v8bf*)(pa);
            v8bf a1 = *(const v8bf*)(pa + 16);
            v16bf av;
            #pragma unroll
            for (int j = 0; j < 8; ++j) { av[j] = a0[j]; av[j + 8] = a1[j]; }

            // B fragments: lane -> col N=l15, K = kk + lhi*8 + {0..7, 16..23}
            const bf16_t* pb0 = sB + ((wn * 2 + 0) * 16 + l15) * PANEL_STRIDE + kk + lhi * 8;
            const bf16_t* pb1 = sB + ((wn * 2 + 1) * 16 + l15) * PANEL_STRIDE + kk + lhi * 8;
            v8bf b00 = *(const v8bf*)(pb0);
            v8bf b01 = *(const v8bf*)(pb0 + 16);
            v8bf b10 = *(const v8bf*)(pb1);
            v8bf b11 = *(const v8bf*)(pb1 + 16);
            v16bf bv0, bv1;
            #pragma unroll
            for (int j = 0; j < 8; ++j) {
                bv0[j] = b00[j]; bv0[j + 8] = b01[j];
                bv1[j] = b10[j]; bv1[j + 8] = b11[j];
            }
            acc0 = __builtin_amdgcn_wmma_f32_16x16x32_bf16(
                false, av, false, bv0, (short)0, acc0, false, false);
            acc1 = __builtin_amdgcn_wmma_f32_16x16x32_bf16(
                false, av, false, bv1, (short)0, acc1, false, false);
        }
    }

    // Epilogue: C row m = bm*64 + wm*16 + lhi*8 + r, col = bn*64 + wn*32 + nt*16 + l15
    const int mrow = bm * 64 + wm * 16 + lhi * 8;
    #pragma unroll
    for (int nt = 0; nt < 2; ++nt) {
        const int col = bn * 64 + wn * 32 + nt * 16 + l15;
        const float bias = b_ih[col] + b_hh[col];
        float* pc = C + (size_t)mrow * G4 + col;
        const v8f* accp = nt ? &acc1 : &acc0;
        #pragma unroll
        for (int r = 0; r < 8; ++r) pc[(size_t)r * G4] = (*accp)[r] + bias;
    }
}

// ---------------------------------------------------------------------------
// Copy initial hidden state into the broadcast buffer
// ---------------------------------------------------------------------------
__global__ void init_h_kernel(const float* __restrict__ h0,
                              float* __restrict__ hcur) {
    int i = threadIdx.x;
    if (i < HID) hcur[i] = h0[i];
}

// ---------------------------------------------------------------------------
// Recurrent scan over T.  Launched as a 16-workgroup cluster (1 WG / WGP).
// WG g owns cells j in [32g, 32g+32) => gate rows {j, j+512, j+1024, j+1536}.
// Its 128 W_hh rows live in LDS (f32, padded stride) for the whole scan.
// ---------------------------------------------------------------------------
#define WSTRIDE (HID + 4)   // 516 floats

__global__ void lstm_scan_kernel(const float* __restrict__ XG,   // T x 2048
                                 const float* __restrict__ Whh,  // 2048 x 512
                                 const float* __restrict__ c0,   // 512
                                 float* __restrict__ hcur,       // 512
                                 bf16_t* __restrict__ Hout) {    // T x 512
    extern __shared__ float smem[];
    float* sW = smem;                       // 128 * WSTRIDE
    float* sh = sW + 128 * WSTRIDE;         // 512
    float* sg = sh + HID;                   // 128 gate pre-activations
    float* sc = sg + 128;                   // 32 cell states

    const int g   = blockIdx.x;             // workgroup-in-cluster 0..15
    const int tid = threadIdx.x;            // 0..255

    for (int idx = tid; idx < 128 * HID; idx += 256) {
        int lr = idx / HID, k = idx - lr * HID;
        int q = lr >> 5, jj = lr & 31;
        int R = q * HID + g * 32 + jj;
        sW[lr * WSTRIDE + k] = Whh[(size_t)R * HID + k];
    }
    if (tid < 32) sc[tid] = c0[g * 32 + tid];
    __syncthreads();

    const int lr   = tid >> 1;              // row 0..127
    const int half = tid & 1;               // k-interleave phase

    for (int t = 0; t < T_STEPS; ++t) {
        // snapshot h[t-1]; prefetch upcoming XG rows while we're at it
        for (int i = tid; i < HID; i += 256) sh[i] = hcur[i];
        if (tid < 32)
            __builtin_prefetch(&XG[(size_t)(t + 4 < T_STEPS ? t + 4 : t) * G4 + g * 32 + tid], 0, 2);
        __syncthreads();
        __builtin_amdgcn_s_cluster_barrier();   // everyone has read h[t-1]

        const float* wrow = sW + lr * WSTRIDE;
        float acc = 0.f;
        #pragma unroll 8
        for (int k = half; k < HID; k += 2)
            acc = fmaf(wrow[k], sh[k], acc);
        acc += __shfl_xor(acc, 1, 32);
        if (half == 0) sg[lr] = acc;
        __syncthreads();

        if (tid < 32) {
            const int jj = tid;
            const size_t xb = (size_t)t * G4 + g * 32 + jj;
            float gi = sg[jj]       + XG[xb];
            float gf = sg[32 + jj]  + XG[xb + 512];
            float gc = sg[64 + jj]  + XG[xb + 1024];
            float go = sg[96 + jj]  + XG[xb + 1536];
            gi = 1.f / (1.f + __expf(-gi));
            gf = 1.f / (1.f + __expf(-gf));
            gc = tanhf(gc);
            go = 1.f / (1.f + __expf(-go));
            float c = gf * sc[jj] + gi * gc;
            sc[jj] = c;
            float h = go * tanhf(c);
            hcur[g * 32 + jj] = h;
            Hout[(size_t)t * HID + g * 32 + jj] = (bf16_t)h;
        }
        __threadfence();
        __syncthreads();
        __builtin_amdgcn_s_cluster_barrier();   // h[t] globally visible
    }
}

// ---------------------------------------------------------------------------
// Head projection + BCE loss, per-block partial sums (deterministic)
// ---------------------------------------------------------------------------
__global__ void head_loss_kernel(const bf16_t* __restrict__ H2,
                                 const float* __restrict__ Whead, // 2 x 512
                                 const float* __restrict__ bhead, // 2
                                 const float* __restrict__ truth, // T x 2
                                 float* __restrict__ partial) {
    __shared__ float red[256];
    const int t = blockIdx.x * blockDim.x + threadIdx.x;
    float local = 0.f;
    if (t < T_STEPS) {
        float l0 = bhead[0], l1 = bhead[1];
        const bf16_t* hp = H2 + (size_t)t * HID;
        #pragma unroll 4
        for (int k = 0; k < HID; ++k) {
            float hv = (float)hp[k];
            l0 = fmaf(hv, Whead[k], l0);
            l1 = fmaf(hv, Whead[HID + k], l1);
        }
        float lg[2] = {l0, l1};
        #pragma unroll
        for (int o = 0; o < 2; ++o) {
            float y  = truth[t * 2 + o];
            float p  = 1.f / (1.f + expf(-lg[o]));
            float lp = fmaxf(logf(p), -100.f);
            float lq = fmaxf(log1pf(-p), -100.f);
            local += y * lp + (1.f - y) * lq;
        }
    }
    red[threadIdx.x] = local;
    __syncthreads();
    for (int s = 128; s > 0; s >>= 1) {
        if (threadIdx.x < s) red[threadIdx.x] += red[threadIdx.x + s];
        __syncthreads();
    }
    if (threadIdx.x == 0) partial[blockIdx.x] = red[0];
}

__global__ void final_reduce_kernel(const float* __restrict__ partial, int n,
                                    float* __restrict__ out) {
    float s = 0.f;
    for (int i = threadIdx.x; i < n; i += 32) s += partial[i];
    #pragma unroll
    for (int off = 16; off; off >>= 1) s += __shfl_xor(s, off, 32);
    if (threadIdx.x == 0) out[0] = -s / (float)(T_STEPS * 2);
}

// ---------------------------------------------------------------------------
extern "C" void kernel_launch(void* const* d_in, const int* in_sizes, int n_in,
                              void* d_out, int out_size, void* d_ws, size_t ws_size,
                              hipStream_t stream) {
    const float* x      = (const float*)d_in[0];
    const float* truth  = (const float*)d_in[1];
    const float* h0     = (const float*)d_in[2];
    const float* c0     = (const float*)d_in[3];
    const float* W_ih0  = (const float*)d_in[4];
    const float* W_hh0  = (const float*)d_in[5];
    const float* b_ih0  = (const float*)d_in[6];
    const float* b_hh0  = (const float*)d_in[7];
    const float* W_ih1  = (const float*)d_in[8];
    const float* W_hh1  = (const float*)d_in[9];
    const float* b_ih1  = (const float*)d_in[10];
    const float* b_hh1  = (const float*)d_in[11];
    const float* W_head = (const float*)d_in[12];
    const float* b_head = (const float*)d_in[13];

    size_t off = 0;
    auto carve = [&](size_t bytes) -> void* {
        void* p = (char*)d_ws + off;
        off += (bytes + 255) & ~(size_t)255;
        return p;
    };
    bf16_t* Xbf    = (bf16_t*)carve((size_t)T_STEPS * HID * 2);  // 8 MB
    bf16_t* Wih0b  = (bf16_t*)carve((size_t)G4 * HID * 2);       // 2 MB
    bf16_t* Wih1b  = (bf16_t*)carve((size_t)G4 * HID * 2);       // 2 MB
    bf16_t* Hseq   = (bf16_t*)carve((size_t)T_STEPS * HID * 2);  // 8 MB
    float*  XG     = (float*) carve((size_t)T_STEPS * G4 * 4);   // 64 MB
    float*  hcur   = (float*) carve(HID * 4);
    float*  part   = (float*) carve(64 * 4);

    const size_t scanLds =
        (size_t)(128 * WSTRIDE + HID + 128 + 32) * sizeof(float); // ~261 KB
    (void)hipFuncSetAttribute((const void*)lstm_scan_kernel,
                              hipFuncAttributeMaxDynamicSharedMemorySize,
                              (int)scanLds);

    const int gemmBlocks = (T_STEPS / 64) * (G4 / 64);  // 4096

    // ---- layer 0 ----
    f32_to_bf16_kernel<<<1024, 256, 0, stream>>>(x,     Xbf,   T_STEPS * HID);
    f32_to_bf16_kernel<<<512,  256, 0, stream>>>(W_ih0, Wih0b, G4 * HID);
    f32_to_bf16_kernel<<<512,  256, 0, stream>>>(W_ih1, Wih1b, G4 * HID);
    gemm_wmma_bf16_kernel<<<gemmBlocks, 256, 0, stream>>>(
        Xbf, Wih0b, b_ih0, b_hh0, XG, T_STEPS);
    init_h_kernel<<<1, 512, 0, stream>>>(h0, hcur);
    lstm_scan_kernel<<<16, 256, scanLds, stream>>>(XG, W_hh0, c0, hcur, Hseq);

    // ---- layer 1 ----
    gemm_wmma_bf16_kernel<<<gemmBlocks, 256, 0, stream>>>(
        Hseq, Wih1b, b_ih1, b_hh1, XG, T_STEPS);
    init_h_kernel<<<1, 512, 0, stream>>>(h0 + HID, hcur);
    lstm_scan_kernel<<<16, 256, scanLds, stream>>>(XG, W_hh1, c0 + HID, hcur, Hseq);

    // ---- head + loss ----
    head_loss_kernel<<<T_STEPS / 256, 256, 0, stream>>>(
        Hseq, W_head, b_head, truth, part);
    final_reduce_kernel<<<1, 32, 0, stream>>>(part, T_STEPS / 256, (float*)d_out);
}
